// SAGELayer_42219528520315
// MI455X (gfx1250) — matrix operand order repaired and checked by reading, not compiled, not run on hardware
//
#include <hip/hip_runtime.h>

#define NROWS    16384
#define FIN      256
#define FOUT     256
#define K2       512      // 2*FIN
#define MBLK     64       // rows per workgroup
#define KC       32
#define NTHREADS 256      // 8 waves = 4 row-strips x 2 column-teams
#define SA_STRIDE 40      // 32 + 8 pad (16B) -> conflict-free A-frag reads
#define HO_STRIDE 520     // 512 + 8 pad

typedef __attribute__((ext_vector_type(8)))  float  v8f;
typedef __attribute__((ext_vector_type(8)))  __bf16 v8bf;
typedef __attribute__((ext_vector_type(16))) __bf16 v16bf;

__device__ __forceinline__ unsigned short cvt_bf16(float f) {
  unsigned int u = __float_as_uint(f);
  u += 0x7FFFu + ((u >> 16) & 1u);          // round-to-nearest-even
  return (unsigned short)(u >> 16);
}

// Build a 16-element bf16 fragment from two 16-byte halves.
__device__ __forceinline__ v16bf ld_frag(const unsigned short* p0,
                                         const unsigned short* p1) {
  v8bf lo = *(const v8bf*)p0;
  v8bf hi = *(const v8bf*)p1;
  return __builtin_shufflevector(lo, hi, 0,1,2,3,4,5,6,7,8,9,10,11,12,13,14,15);
}

// Issue global loads for one 64x32 adj tile (kept in regs, latency overlapped).
__device__ __forceinline__ void tile_load(float4* stg, const float* __restrict__ adj,
                                          int r0, int k0, int tid) {
  #pragma unroll
  for (int i = 0; i < 2; ++i) {
    int f  = tid + i * NTHREADS;          // float4 index 0..511
    int r  = f >> 3;
    int k4 = f & 7;
    stg[i] = *((const float4*)(adj + (size_t)(r0 + r) * NROWS + k0) + k4);
  }
}

// Convert the staged regs to bf16 and store into an LDS A-tile buffer.
__device__ __forceinline__ void tile_store(unsigned short* buf, const float4* stg, int tid) {
  #pragma unroll
  for (int i = 0; i < 2; ++i) {
    int f  = tid + i * NTHREADS;
    int r  = f >> 3;
    int k4 = f & 7;
    float4 v = stg[i];
    unsigned int p0 = (unsigned int)cvt_bf16(v.x) | ((unsigned int)cvt_bf16(v.y) << 16);
    unsigned int p1 = (unsigned int)cvt_bf16(v.z) | ((unsigned int)cvt_bf16(v.w) << 16);
    *(uint2*)(&buf[r * SA_STRIDE + k4 * 4]) = make_uint2(p0, p1);
  }
}

// xT[c][k] = bf16(x[k][c]) : [FIN][NROWS] bf16, K-contiguous for B-fragments.
__global__ void cvt_x_kernel(const float* __restrict__ x,
                             unsigned short* __restrict__ xT) {
  int t = blockIdx.x * blockDim.x + threadIdx.x;   // 4M threads
  int c = t & (FIN - 1);
  int k = t >> 8;
  xT[(size_t)c * NROWS + k] = cvt_bf16(x[(size_t)k * FIN + c]);
}

// wT[n][k] = bf16(w[k][n]) : [FOUT][K2] bf16
__global__ void cvt_w_kernel(const float* __restrict__ w,
                             unsigned short* __restrict__ wT) {
  int t = blockIdx.x * blockDim.x + threadIdx.x;   // 131072 threads
  int n = t & (FOUT - 1);
  int k = t >> 8;
  wT[(size_t)n * K2 + k] = cvt_bf16(w[(size_t)k * FOUT + n]);
}

__global__ void __launch_bounds__(NTHREADS)
sage_fused_kernel(const float* __restrict__ x,
                  const float* __restrict__ adj,
                  const unsigned short* __restrict__ xT,
                  const unsigned short* __restrict__ wT,
                  float* __restrict__ out) {
  extern __shared__ char smem[];
  unsigned short* hOld = (unsigned short*)smem;           // [MBLK][HO_STRIDE] bf16
  unsigned short* sA0  = hOld + MBLK * HO_STRIDE;         // [MBLK][SA_STRIDE] bf16
  unsigned short* sA1  = sA0 + MBLK * SA_STRIDE;          // double buffer
  float* partSS        = (float*)sA0;                     // reused post-GEMM (512 B)

  const int tid  = threadIdx.x;
  const int wave = tid >> 5;
  const int lane = tid & 31;
  const int lh   = lane >> 4;       // half-wave (K-offset select)
  const int lm   = lane & 15;       // row (A) / col (B) within tile
  const int roww = wave & 3;        // row strip: rows [roww*16, +16)
  const int team = wave >> 2;       // column team: n-tiles [team*8, +8)
  const int r0   = blockIdx.x * MBLK;

  // Stage self-features x[r0..r0+63][:] as bf16 into h_old cols [0,256).
  for (int i = 0; i < MBLK; ++i)
    hOld[i * HO_STRIDE + tid] = cvt_bf16(x[(size_t)(r0 + i) * FIN + tid]);

  // ---------------- Phase 1: h_neigh = adj @ x  (K = 16384) ----------------
  v8f acc[8] = {};   // wave: 16 rows x 128 cols (8 tiles), f32 accumulators

  float4 stg[2];
  tile_load(stg, adj, r0, 0, tid);    // prologue: tile 0
  tile_store(sA0, stg, tid);
  __syncthreads();

  unsigned short* cur = sA0;
  unsigned short* nxt = sA1;
  const int NT = NROWS / KC;          // 512 K-chunks

  #pragma unroll 1
  for (int t = 0; t < NT; ++t) {
    const int k0 = t * KC;

    // Kick off next adj tile's global loads NOW so HBM latency hides
    // behind the WMMAs below.
    if (t + 1 < NT) {
      tile_load(stg, adj, r0, k0 + KC, tid);
      if (t + 2 < NT)   // speculative prefetch two tiles ahead
        __builtin_prefetch(adj + (size_t)(r0 + (tid >> 2)) * NROWS + (k0 + 2 * KC) + (tid & 3) * 8, 0, 0);
    }

    // A fragment: row = roww*16 + lm, K contiguous at 16*lh .. 16*lh+15
    const unsigned short* ap = &cur[(roww * 16 + lm) * SA_STRIDE + 16 * lh];
    v16bf afrag = ld_frag(ap, ap + 8);

    // All 8 B fragments live at once -> 16 global_load_b128 clustered,
    // waits staggered instead of one s_wait_loadcnt 0 per WMMA.
    v16bf bfr[8];
    #pragma unroll
    for (int j = 0; j < 8; ++j) {
      int nt = team * 8 + j;
      const unsigned short* bp = xT + (size_t)(nt * 16 + lm) * NROWS + k0 + 16 * lh;
      bfr[j] = ld_frag(bp, bp + 8);
    }
    #pragma unroll
    for (int j = 0; j < 8; ++j)
      acc[j] = __builtin_amdgcn_wmma_f32_16x16x32_bf16(
          false, afrag, false, bfr[j], (short)0, acc[j], false, false);

    // Write next tile into the other LDS buffer; single barrier per iter.
    if (t + 1 < NT) tile_store(nxt, stg, tid);
    __syncthreads();

    unsigned short* tmp = cur; cur = nxt; nxt = tmp;
  }

  // Spill h_neigh strip into h_old cols [256,512) as bf16.
  #pragma unroll
  for (int j = 0; j < 8; ++j)
    #pragma unroll
    for (int v = 0; v < 8; ++v)
      hOld[(roww * 16 + v + 8 * lh) * HO_STRIDE + (FIN + (team * 8 + j) * 16 + lm)] =
          cvt_bf16(acc[j][v]);
  __syncthreads();

  // ---------------- Phase 2: h_new = sigmoid(h_old @ w)  (K = 512) ---------
  v8f acc2[8] = {};
  #pragma unroll 1
  for (int k0 = 0; k0 < K2; k0 += KC) {
    const unsigned short* ap2 = &hOld[(roww * 16 + lm) * HO_STRIDE + k0 + 16 * lh];
    v16bf afrag = ld_frag(ap2, ap2 + 8);
    v16bf bfr[8];
    #pragma unroll
    for (int j = 0; j < 8; ++j) {
      int nt = team * 8 + j;
      const unsigned short* bp = wT + (size_t)(nt * 16 + lm) * K2 + k0 + 16 * lh;
      bfr[j] = ld_frag(bp, bp + 8);
    }
    #pragma unroll
    for (int j = 0; j < 8; ++j)
      acc2[j] = __builtin_amdgcn_wmma_f32_16x16x32_bf16(
          false, afrag, false, bfr[j], (short)0, acc2[j], false, false);
  }

  // Sigmoid
  #pragma unroll
  for (int j = 0; j < 8; ++j)
    #pragma unroll
    for (int v = 0; v < 8; ++v)
      acc2[j][v] = 1.0f / (1.0f + __expf(-acc2[j][v]));

  // Row-wise L2 normalize. Each row's 256 cols are split across the two
  // column teams -> shuffle-reduce within the wave, then combine the two
  // team partials through LDS.
  #pragma unroll
  for (int v = 0; v < 8; ++v) {
    float ss = 0.0f;
    #pragma unroll
    for (int j = 0; j < 8; ++j) { float g = acc2[j][v]; ss += g * g; }
    ss += __shfl_xor(ss, 1);
    ss += __shfl_xor(ss, 2);
    ss += __shfl_xor(ss, 4);
    ss += __shfl_xor(ss, 8);
    int rloc = roww * 16 + v + 8 * lh;
    if (lm == 0) partSS[rloc * 2 + team] = ss;
  }
  __syncthreads();

  #pragma unroll
  for (int v = 0; v < 8; ++v) {
    int rloc  = roww * 16 + v + 8 * lh;
    float tot = partSS[rloc * 2 + 0] + partSS[rloc * 2 + 1];
    float inv = 1.0f / sqrtf(tot);
    float* orow = out + (size_t)(r0 + rloc) * FOUT;
    #pragma unroll
    for (int j = 0; j < 8; ++j)
      orow[(team * 8 + j) * 16 + lm] = acc2[j][v] * inv;
  }
}

extern "C" void kernel_launch(void* const* d_in, const int* in_sizes, int n_in,
                              void* d_out, int out_size, void* d_ws, size_t ws_size,
                              hipStream_t stream) {
  (void)in_sizes; (void)n_in; (void)out_size; (void)ws_size;
  const float* x   = (const float*)d_in[0];
  const float* adj = (const float*)d_in[1];
  const float* w   = (const float*)d_in[2];
  float* out = (float*)d_out;

  unsigned short* xT = (unsigned short*)d_ws;            // 8 MB
  unsigned short* wT = xT + (size_t)FIN * NROWS;         // +256 KB

  cvt_x_kernel<<<(NROWS * FIN) / NTHREADS, NTHREADS, 0, stream>>>(x, xT);
  cvt_w_kernel<<<(K2 * FOUT) / NTHREADS, NTHREADS, 0, stream>>>(w, wT);

  size_t shmem = (size_t)(MBLK * HO_STRIDE + 2 * MBLK * SA_STRIDE) * sizeof(unsigned short); // 75 KB
  (void)hipFuncSetAttribute((const void*)sage_fused_kernel,
                            hipFuncAttributeMaxDynamicSharedMemorySize, (int)shmem);
  sage_fused_kernel<<<NROWS / MBLK, NTHREADS, shmem, stream>>>(x, adj, xT, wT, out);
}
